// SimVQCodebook_14946486190088
// MI455X (gfx1250) — matrix operand order repaired and checked by reading, not compiled, not run on hardware
//
#include <hip/hip_runtime.h>
#include <stdint.h>

#define B_ROWS 32768
#define K_ROWS 8192
#define DIM    256
#define CHUNK  32   // codebook entries staged per LDS buffer (16 KB)

typedef __attribute__((ext_vector_type(16))) __bf16        v16bf;
typedef __attribute__((ext_vector_type(8)))  __bf16        bf16x8;
typedef __attribute__((ext_vector_type(8)))  float         v8f;
typedef __attribute__((ext_vector_type(4)))  float         f32x4;
typedef __attribute__((ext_vector_type(4)))  unsigned short u16x4;

static __device__ __forceinline__ unsigned short f2bf(float f) {
  union { float f; unsigned int u; } v; v.f = f;
  unsigned int r = v.u + 0x7FFFu + ((v.u >> 16) & 1u);   // round-to-nearest-even
  return (unsigned short)(r >> 16);
}

// Async global -> LDS 128-bit copy (gfx1250, tracked by ASYNCcnt).
// INST_OFFSET is added to both the LDS destination and the global source.
#define ASYNC_LD16(ldsaddr, gaddr, OFF)                                        \
  asm volatile("global_load_async_to_lds_b128 %0, %1, off offset:" OFF         \
               :: "v"(ldsaddr), "v"(gaddr) : "memory")

static __device__ __forceinline__ void wait_async0() {
#if __has_builtin(__builtin_amdgcn_s_wait_asynccnt)
  __builtin_amdgcn_s_wait_asynccnt(0);
#else
  asm volatile("s_wait_asynccnt 0x0" ::: "memory");
#endif
}

// ---------------- z -> bf16 ----------------
__global__ void k_zbf16(const float* __restrict__ z, unsigned short* __restrict__ zb) {
  int t = blockIdx.x * blockDim.x + threadIdx.x;   // one float4 per thread
  f32x4 v = ((const f32x4*)z)[t];
  u16x4 o;
  o.x = f2bf(v.x); o.y = f2bf(v.y); o.z = f2bf(v.z); o.w = f2bf(v.w);
  ((u16x4*)zb)[t] = o;
}

// ---------------- qc = codebook @ W^T (fp32, 64x64 tile) ----------------
#define QT 64
#define QD 32
__global__ __launch_bounds__(256) void k_qc(const float* __restrict__ Cb,
                                            const float* __restrict__ W,
                                            float* __restrict__ qc) {
  __shared__ float sA[QT][QD + 1];   // codebook tile [k][d]
  __shared__ float sB[QT][QD + 1];   // W tile        [e][d]
  const int tid = threadIdx.x;
  const int tx = tid & 15, ty = tid >> 4;
  const int k0 = blockIdx.y * QT, e0 = blockIdx.x * QT;
  float acc[4][4] = {};
  for (int dc = 0; dc < DIM; dc += QD) {
#pragma unroll
    for (int i = 0; i < 2; ++i) {
      int f = tid + i * 256;        // 512 float4s per tile
      int row = f >> 3;             // 8 float4s per 32-wide row
      int c4  = (f & 7) * 4;
      f32x4 a = *(const f32x4*)(Cb + (size_t)(k0 + row) * DIM + dc + c4);
      f32x4 b = *(const f32x4*)(W  + (size_t)(e0 + row) * DIM + dc + c4);
      sA[row][c4 + 0] = a.x; sA[row][c4 + 1] = a.y; sA[row][c4 + 2] = a.z; sA[row][c4 + 3] = a.w;
      sB[row][c4 + 0] = b.x; sB[row][c4 + 1] = b.y; sB[row][c4 + 2] = b.z; sB[row][c4 + 3] = b.w;
    }
    __syncthreads();
#pragma unroll 4
    for (int d = 0; d < QD; ++d) {
      float ar[4], br[4];
#pragma unroll
      for (int i = 0; i < 4; ++i) { ar[i] = sA[ty * 4 + i][d]; br[i] = sB[tx * 4 + i][d]; }
#pragma unroll
      for (int i = 0; i < 4; ++i)
#pragma unroll
        for (int j = 0; j < 4; ++j) acc[i][j] = fmaf(ar[i], br[j], acc[i][j]);
    }
    __syncthreads();
  }
#pragma unroll
  for (int i = 0; i < 4; ++i)
#pragma unroll
    for (int j = 0; j < 4; ++j)
      qc[(size_t)(k0 + ty * 4 + i) * DIM + e0 + tx * 4 + j] = acc[i][j];
}

// ---------------- c_sq + qc -> bf16 ----------------
__global__ void k_csq(const float* __restrict__ qc, unsigned short* __restrict__ qcb,
                      float* __restrict__ csq) {
  const int row = blockIdx.x;
  const int t = threadIdx.x;   // 64 threads, 4 elems each
  f32x4 v = *(const f32x4*)(qc + (size_t)row * DIM + t * 4);
  u16x4 o;
  o.x = f2bf(v.x); o.y = f2bf(v.y); o.z = f2bf(v.z); o.w = f2bf(v.w);
  *((u16x4*)(qcb + (size_t)row * DIM) + t) = o;
  float s = v.x * v.x + v.y * v.y + v.z * v.z + v.w * v.w;
  __shared__ float red[64];
  red[t] = s; __syncthreads();
  for (int off = 32; off > 0; off >>= 1) {
    if (t < off) red[t] += red[t + off];
    __syncthreads();
  }
  if (t == 0) csq[row] = red[0];
}

// ---------------- main: fused bf16-WMMA distance GEMM + argmin ----------------
__global__ __launch_bounds__(128) void k_main(const unsigned short* __restrict__ zb,
                                              const unsigned short* __restrict__ qcb,
                                              const float* __restrict__ csq,
                                              int* __restrict__ idxout,
                                              float* __restrict__ fidxout) {
  __shared__ __align__(16) unsigned char smem[2 * CHUNK * DIM * 2];   // 32 KB, 2 buffers
  unsigned short* stage = (unsigned short*)smem;

  const int tid = threadIdx.x;
  const int wave = tid >> 5, lane = tid & 31;
  const int lane16 = lane & 15, hi = lane >> 4;
  const int wgRow0 = blockIdx.x * 128;
  const int rbase = wgRow0 + wave * 32;

  // Persistent A fragments: 2 M-tiles x 8 K-steps.
  // 16-bit A layout: lanes 0-15 row M=lane, K in [0..7] U [16..23];
  // lanes 16-31 same rows, K in [8..15] U [24..31].
  v16bf a[2][8];
#pragma unroll
  for (int mt = 0; mt < 2; ++mt) {
    const unsigned short* zr = zb + (size_t)(rbase + mt * 16 + lane16) * DIM;
#pragma unroll
    for (int s = 0; s < 8; ++s) {
      const int base = s * 32 + hi * 8;
      bf16x8 lo = *(const bf16x8*)(zr + base);
      bf16x8 hh = *(const bf16x8*)(zr + base + 16);
      v16bf t;
#pragma unroll
      for (int i = 0; i < 8; ++i) { t[i] = lo[i]; t[8 + i] = hh[i]; }
      a[mt][s] = t;
    }
  }

  float best[2][8];
  int   bidx[2][8];
#pragma unroll
  for (int mt = 0; mt < 2; ++mt)
#pragma unroll
    for (int i = 0; i < 8; ++i) { best[mt][i] = 3.4e38f; bidx[mt][i] = 0; }

  const int nChunks = K_ROWS / CHUNK;   // 256
  const uint32_t ldsBase = (uint32_t)(uintptr_t)smem;   // LDS byte address of buffer 0
  const uint64_t gBase   = (uint64_t)(uintptr_t)qcb;

  // Prologue: async-stage chunk 0 into buffer 0 (each thread copies 128 B)
  {
    const uint64_t g = gBase + (uint64_t)tid * 16;
    const uint32_t l = ldsBase + (uint32_t)tid * 16;
    ASYNC_LD16(l, g, "0");     ASYNC_LD16(l, g, "2048");
    ASYNC_LD16(l, g, "4096");  ASYNC_LD16(l, g, "6144");
    ASYNC_LD16(l, g, "8192");  ASYNC_LD16(l, g, "10240");
    ASYNC_LD16(l, g, "12288"); ASYNC_LD16(l, g, "14336");
    wait_async0();
  }
  __syncthreads();

  for (int c = 0; c < nChunks; ++c) {
    const int cur = c & 1;

    // Kick off async DMA of the next chunk into the other buffer;
    // it overlaps with this chunk's 32 WMMAs.
    if (c + 1 < nChunks) {
      const uint64_t g = gBase + (uint64_t)(c + 1) * (CHUNK * DIM * 2) + (uint64_t)tid * 16;
      const uint32_t l = ldsBase + (uint32_t)((cur ^ 1) * (CHUNK * DIM * 2) + tid * 16);
      ASYNC_LD16(l, g, "0");     ASYNC_LD16(l, g, "2048");
      ASYNC_LD16(l, g, "4096");  ASYNC_LD16(l, g, "6144");
      ASYNC_LD16(l, g, "8192");  ASYNC_LD16(l, g, "10240");
      ASYNC_LD16(l, g, "12288"); ASYNC_LD16(l, g, "14336");
    }

    const unsigned short* sb = stage + cur * CHUNK * DIM;
#pragma unroll
    for (int nt = 0; nt < 2; ++nt) {
      const int kk = c * CHUNK + nt * 16 + lane16;
      const float cs = csq[kk];
      // B layout (32x16 bf16): column N=lane&15; lanes 0-15 hold K [0..7] U [16..23],
      // lanes 16-31 hold K [8..15] U [24..31] -> two 16B runs per lane.
      const unsigned short* bp = sb + (nt * 16 + lane16) * DIM + hi * 8;
      v8f acc0, acc1;
#pragma unroll
      for (int i = 0; i < 8; ++i) { acc0[i] = 0.f; acc1[i] = 0.f; }
      v16bf bcur = __builtin_shufflevector(*(const bf16x8*)bp, *(const bf16x8*)(bp + 16), 0, 1, 2, 3, 4, 5, 6, 7, 8, 9, 10, 11, 12, 13, 14, 15);   // s = 0
#pragma unroll
      for (int s = 0; s < 8; ++s) {
        v16bf bnext;
        if (s < 7) bnext = __builtin_shufflevector(*(const bf16x8*)(bp + (s + 1) * 32), *(const bf16x8*)(bp + (s + 1) * 32 + 16), 0, 1, 2, 3, 4, 5, 6, 7, 8, 9, 10, 11, 12, 13, 14, 15);
        acc0 = __builtin_amdgcn_wmma_f32_16x16x32_bf16(false, a[0][s], false, bcur,
                                                       (short)0, acc0, false, false);
        acc1 = __builtin_amdgcn_wmma_f32_16x16x32_bf16(false, a[1][s], false, bcur,
                                                       (short)0, acc1, false, false);
        bcur = bnext;
      }
#pragma unroll
      for (int i = 0; i < 8; ++i) {
        float s0 = fmaf(-2.f, acc0[i], cs);
        if (s0 < best[0][i]) { best[0][i] = s0; bidx[0][i] = kk; }
        float s1 = fmaf(-2.f, acc1[i], cs);
        if (s1 < best[1][i]) { best[1][i] = s1; bidx[1][i] = kk; }
      }
    }

    // Next chunk's DMA must be complete (and everyone done reading) before
    // the buffers swap roles.
    wait_async0();
    __syncthreads();
  }

  // Cross-lane argmin: each lane holds best over k = lane16 (mod 16)
  // for rows M = i + 8*hi within its tiles.
  struct Pair { float s; int k; };
  Pair* red = (Pair*)smem;   // 128 rows x 16 slots = 16 KB
#pragma unroll
  for (int mt = 0; mt < 2; ++mt)
#pragma unroll
    for (int i = 0; i < 8; ++i) {
      int row_local = wave * 32 + mt * 16 + i + hi * 8;
      Pair p; p.s = best[mt][i]; p.k = bidx[mt][i];
      red[row_local * 16 + lane16] = p;
    }
  __syncthreads();
  {
    const int row_local = tid;       // 0..127
    Pair p = red[row_local * 16];
    for (int s = 1; s < 16; ++s) {
      Pair q = red[row_local * 16 + s];
      if (q.s < p.s || (q.s == p.s && q.k < p.k)) p = q;
    }
    const int b = wgRow0 + row_local;
    idxout[b] = p.k;
    fidxout[b] = (float)p.k;
  }
}

// ---------------- gather quantized rows ----------------
__global__ void k_gather(const float* __restrict__ qc, const int* __restrict__ idx,
                         float* __restrict__ out) {
  const int t = threadIdx.x;
  const int row = blockIdx.x * 4 + (t >> 6);
  const int c = (t & 63) * 4;
  const int k = idx[row];
  f32x4 v = *(const f32x4*)(qc + (size_t)k * DIM + c);
  *(f32x4*)(out + (size_t)row * DIM + c) = v;
}

extern "C" void kernel_launch(void* const* d_in, const int* in_sizes, int n_in,
                              void* d_out, int out_size, void* d_ws, size_t ws_size,
                              hipStream_t stream) {
  const float* z  = (const float*)d_in[0];
  const float* cb = (const float*)d_in[1];
  const float* W  = (const float*)d_in[2];
  float* out = (float*)d_out;

  char* ws = (char*)d_ws;
  float*          qc  = (float*)(ws + 0);                        // 8 MB
  unsigned short* qcb = (unsigned short*)(ws + 8 * 1024 * 1024); // 4 MB
  unsigned short* zbf = (unsigned short*)(ws + 12 * 1024 * 1024);// 16 MB
  float*          csq = (float*)(ws + 28 * 1024 * 1024);         // 32 KB
  int*            idx = (int*)(ws + 28 * 1024 * 1024 + 64 * 1024);

  k_zbf16 <<<(B_ROWS * DIM / 4) / 256, 256, 0, stream>>>(z, zbf);
  k_qc    <<<dim3(DIM / QT, K_ROWS / QT), 256, 0, stream>>>(cb, W, qc);
  k_csq   <<<K_ROWS, 64, 0, stream>>>(qc, qcb, csq);
  k_main  <<<B_ROWS / 128, 128, 0, stream>>>(zbf, qcb, csq, idx,
                                             out + (size_t)B_ROWS * DIM);
  k_gather<<<B_ROWS / 4, 256, 0, stream>>>(qc, idx, out);
}
